// Tagger_88356067213740
// MI455X (gfx1250) — matrix operand order repaired and checked by reading, not compiled, not run on hardware
//
#include <hip/hip_runtime.h>
#include <hip/hip_bf16.h>
#include <math.h>

// ---------------- types ----------------
typedef __attribute__((ext_vector_type(16))) _Float16      v16h;
typedef __attribute__((ext_vector_type(8)))  float         v8f;
typedef __attribute__((ext_vector_type(4)))  unsigned int  u32x4;
typedef __attribute__((ext_vector_type(4)))  int           v4i;

union AFrag { v16h v; u32x4 q[2]; };
union CFr   { v8f  f; u32x4 q[2]; };

#define AS1 __attribute__((address_space(1)))
#define AS3 __attribute__((address_space(3)))

// ---------------- problem constants ----------------
#define NE   300      // embed dim
#define NEP  320      // embed dim padded to multiple of 32
#define NH   256      // per-direction hidden
#define G4   1024     // 4*NH  (gate width)
#define NK   20       // tagset
#define NB   32       // batch
#define NT   128      // time
#define TB   (NT*NB)  // 4096 rows
#define START_TAG 18
#define STOP_TAG  19
#define NEGF (-10000.0f)

__device__ __forceinline__ float sigm(float x) { return 1.0f / (1.0f + __expf(-x)); }

__device__ __forceinline__ v8f wmma16(const AFrag& a, const AFrag& b, v8f c) {
  // D = A(16x32 f16) * B(32x16 f16) + C(16x16 f32)
  return __builtin_amdgcn_wmma_f32_16x16x32_f16(
      /*neg_a=*/false, a.v, /*neg_b=*/false, b.v,
      /*c_mod=*/(short)0, c, /*reuse_a=*/false, /*reuse_b=*/false);
}

// ============ K0: weight conversion / packing to f16 ============
// wih16: [2][1024][320] (E padded with zeros)
// whh16: [2][1024][256]
// wout16: [32][512]     (rows 20..31 zero)
__global__ void k_convert(const float* __restrict__ wihf, const float* __restrict__ wihb,
                          const float* __restrict__ whhf, const float* __restrict__ whhb,
                          const float* __restrict__ wout,
                          _Float16* __restrict__ wih16, _Float16* __restrict__ whh16,
                          _Float16* __restrict__ wout16) {
  int tid = blockIdx.x * 256 + threadIdx.x;
  const int nWih = 2 * G4 * NEP;   // 655360
  const int nWhh = 2 * G4 * NH;    // 524288
  const int nWo  = 32 * 512;       // 16384
  if (tid < nWih) {
    int dir = tid / (G4 * NEP);
    int r   = tid % (G4 * NEP);
    int n = r / NEP, k = r % NEP;
    const float* src = dir ? wihb : wihf;
    float v = (k < NE) ? src[n * NE + k] : 0.0f;
    wih16[tid] = (_Float16)v;
  } else if (tid < nWih + nWhh) {
    int t2  = tid - nWih;
    int dir = t2 / (G4 * NH);
    int r   = t2 % (G4 * NH);
    const float* src = dir ? whhb : whhf;
    whh16[t2] = (_Float16)src[r];
  } else if (tid < nWih + nWhh + nWo) {
    int t2 = tid - nWih - nWhh;
    int n = t2 / 512, k = t2 % 512;
    float v = (n < NK) ? wout[n * 512 + k] : 0.0f;
    wout16[t2] = (_Float16)v;
  }
}

// ============ K1: embedding gather + f16 convert ============
// x16: [T*B][320] time-major rows (m = t*B + b)
__global__ void k_embed(const int* __restrict__ sent, const float* __restrict__ wemb,
                        _Float16* __restrict__ x16) {
  int tid = blockIdx.x * 256 + threadIdx.x;   // over TB*NEP
  if (tid >= TB * NEP) return;
  int m = tid / NEP, e = tid % NEP;
  int t = m / NB, b = m % NB;
  int tok = sent[b * NT + t];
  float v = (e < NE) ? wemb[(size_t)tok * NE + e] : 0.0f;
  x16[tid] = (_Float16)v;
}

// ============ K2: input projection pre-gates (WMMA GEMM) ============
// pre2[dir][t][n][b]  (gate-transposed, fragment-native for K3):
//   pre2[((dir*NT+t)*G4 + n)*NB + b] = x16[t*B+b,:] . wih16[dir][n,:] + bias[n]
// One wave per (dir, mtile, npair of 2 ntiles). 16384 waves, 8 waves/block.
__global__ void k_pregate(const _Float16* __restrict__ x16, const _Float16* __restrict__ wih16,
                          const float* __restrict__ bf, const float* __restrict__ bb,
                          float* __restrict__ pre2) {
  int lane = threadIdx.x & 31;
  int widx = threadIdx.x >> 5;
  int gw = blockIdx.x * 8 + widx;        // 0..16383
  int dir   = gw >> 13;
  int rem   = gw & 8191;
  int mtile = rem >> 5;                  // 0..255 (16 rows of m = t*B + b)
  int n0    = (rem & 31) * 32;           // 0..992
  int lnlo = lane & 15;
  int hi   = lane >> 4;

  const _Float16* wih = wih16 + (size_t)dir * G4 * NEP;
  v8f c0 = {}, c1 = {};
  int arow = mtile * 16 + lnlo;
  for (int kb = 0; kb < NEP; kb += 32) {
    AFrag a, b0, b1;
    int kA = kb + (hi ? 8 : 0);
    a.q[0] = *(const u32x4*)(x16 + (size_t)arow * NEP + kA);
    a.q[1] = *(const u32x4*)(x16 + (size_t)arow * NEP + kA + 16);
    int kB = kb + (hi ? 8 : 0);
    const u32x4* bp0 = (const u32x4*)(wih + (size_t)(n0 + lnlo) * NEP + kB);
    b0.q[0] = bp0[0]; b0.q[1] = bp0[2];
    const u32x4* bp1 = (const u32x4*)(wih + (size_t)(n0 + 16 + lnlo) * NEP + kB);
    b1.q[0] = bp1[0]; b1.q[1] = bp1[2];
    c0 = wmma16(a, b0, c0);
    c1 = wmma16(a, b1, c1);
  }
  const float* bias = dir ? bb : bf;
  // rows m = mtile*16 .. +15 all share t = mtile>>1; b = (mtile&1)*16 + hi*8 + r
  int t  = mtile >> 1;
  int bb0 = (mtile & 1) * 16 + hi * 8;
  for (int nt = 0; nt < 2; ++nt) {
    int n = n0 + nt * 16 + lnlo;
    float bn = bias[n];
    CFr c; c.f = nt ? c1 : c0;
    for (int r = 0; r < 8; ++r) c.f[r] += bn;
    float* dst = pre2 + (((size_t)(dir * NT + t) * G4 + n) * NB + bb0);
    *(u32x4*)(dst)     = c.q[0];
    *(u32x4*)(dst + 4) = c.q[1];
  }
}

// ============ K3: persistent LSTM recurrence (2 blocks = 2 directions) ============
// 512 threads = 16 waves. Wave w owns hidden slice j in [16w, 16w+16).
// Per step: gates[b, {j, H+j, 2H+j, 3H+j}] = pre2[t,.,b] + h @ W_hh.T  via WMMA.
// c stays in wave registers for all 128 steps; h round-trips through 16KB LDS.
__global__ void __launch_bounds__(512) k_lstm(const float* __restrict__ pre2,
                                              const _Float16* __restrict__ whh16,
                                              const float* __restrict__ h0,
                                              const float* __restrict__ c0g,
                                              _Float16* __restrict__ hs) {
  int dir  = blockIdx.x;
  int lane = threadIdx.x & 31;
  int w    = threadIdx.x >> 5;       // 0..15
  int lnlo = lane & 15, hi = lane >> 4;
  int jb   = w * 16;

  __shared__ __align__(16) _Float16 hsh[NB * NH];   // 16 KB

  for (int idx = threadIdx.x; idx < NB * NH; idx += 512)
    hsh[idx] = (_Float16)h0[(size_t)dir * NB * NH + idx];

  float cc[2][8];
  for (int mt = 0; mt < 2; ++mt)
    for (int r = 0; r < 8; ++r) {
      int b = mt * 16 + r + hi * 8;
      cc[mt][r] = c0g[(size_t)dir * NB * NH + (size_t)b * NH + jb + lnlo];
    }
  __syncthreads();

  const _Float16* whh = whh16 + (size_t)dir * G4 * NH;
  const float*    prd = pre2  + (size_t)dir * NT * G4 * NB;
  _Float16*       hso = hs    + (size_t)dir * NT * NB * NH;

  int nbase[4] = { jb, NH + jb, 2 * NH + jb, 3 * NH + jb };

  for (int step = 0; step < NT; ++step) {
    int tg = dir ? (NT - 1 - step) : step;

    // C init = pre-gates (fragment-native layout: 2x b128 per fragment)
    CFr cf[4][2];
    for (int g = 0; g < 4; ++g)
      for (int mt = 0; mt < 2; ++mt) {
        const float* p = prd + (((size_t)tg * G4 + nbase[g] + lnlo) * NB + mt * 16 + hi * 8);
        cf[g][mt].q[0] = *(const u32x4*)(p);
        cf[g][mt].q[1] = *(const u32x4*)(p + 4);
      }

    // recurrent GEMM: h(32x256) @ W_hh.T slice -> 8 WMMA per k-tile
    for (int kb = 0; kb < NH; kb += 32) {
      AFrag a[2], bfr[4];
      int kA = kb + (hi ? 8 : 0);
      for (int mt = 0; mt < 2; ++mt) {
        int row = mt * 16 + lnlo;
        a[mt].q[0] = *(const u32x4*)(hsh + row * NH + kA);
        a[mt].q[1] = *(const u32x4*)(hsh + row * NH + kA + 16);
      }
      int kB = kb + (hi ? 8 : 0);
      for (int g = 0; g < 4; ++g) {
        const u32x4* bp = (const u32x4*)(whh + (size_t)(nbase[g] + lnlo) * NH + kB);
        bfr[g].q[0] = bp[0]; bfr[g].q[1] = bp[2];
      }
      if (kb + 32 < NH)  // hide next W_hh tile fetch latency
        __builtin_prefetch(whh + (size_t)(nbase[0] + lnlo) * NH + kb + 32, 0, 1);
      for (int g = 0; g < 4; ++g)
        for (int mt = 0; mt < 2; ++mt)
          cf[g][mt].f = wmma16(a[mt], bfr[g], cf[g][mt].f);
    }

    __syncthreads();   // all waves finished reading previous h from LDS

    // LSTM cell elementwise in registers; publish new h
    for (int mt = 0; mt < 2; ++mt)
      for (int r = 0; r < 8; ++r) {
        float iv = sigm(cf[0][mt].f[r]);
        float fv = sigm(cf[1][mt].f[r]);
        float gv = tanhf(cf[2][mt].f[r]);
        float ov = sigm(cf[3][mt].f[r]);
        float c  = fv * cc[mt][r] + iv * gv;
        cc[mt][r] = c;
        float h = ov * tanhf(c);
        int b = mt * 16 + r + hi * 8;
        int j = jb + lnlo;
        _Float16 hh = (_Float16)h;
        hsh[b * NH + j] = hh;
        hso[((size_t)tg * NB + b) * NH + j] = hh;   // sequence output for feats GEMM
      }
    __syncthreads();
  }
}

// ============ K4: output projection (WMMA GEMM, N padded 20->32) ============
// W_out (32KB f16) is block-invariant: stage into LDS with the gfx1250 async
// copy path (global_load_async_to_lds_b128 + s_wait_asynccnt) when available.
// feats[b][t][k] = hcat[m=t*B+b,:512] . wout[k,:] + b_out[k]
__global__ void k_feats(const _Float16* __restrict__ hs, const _Float16* __restrict__ wout16,
                        const float* __restrict__ bout, float* __restrict__ feats) {
  int lane = threadIdx.x & 31;
  int widx = threadIdx.x >> 5;
  int mtile = blockIdx.x * 8 + widx;   // 0..255
  int lnlo = lane & 15, hi = lane >> 4;
  const _Float16* hbk = hs + (size_t)NT * NB * NH;  // backward halves

  __shared__ __align__(16) _Float16 wsh[32 * 512];  // 32 KB staged W_out

#if __has_builtin(__builtin_amdgcn_global_load_async_to_lds_b128)
  for (int i = threadIdx.x; i < (32 * 512) / 8; i += 256) {
    // builtin signature (probe-confirmed): (AS1 v4i* src, AS3 v4i* dst, imm off, imm cpol)
    __builtin_amdgcn_global_load_async_to_lds_b128(
        (AS1 v4i*)(v4i*)(void*)(const void*)(wout16 + (size_t)i * 8),
        (AS3 v4i*)(v4i*)(void*)(wsh + (size_t)i * 8), 0, 0);
  }
#if __has_builtin(__builtin_amdgcn_s_wait_asynccnt)
  __builtin_amdgcn_s_wait_asynccnt(0);
#endif
#else
  for (int i = threadIdx.x; i < 32 * 512; i += 256)
    wsh[i] = wout16[i];
#endif
  __syncthreads();

  v8f c0 = {}, c1 = {};
  int arow = mtile * 16 + lnlo;
  for (int kb = 0; kb < 2 * NH; kb += 32) {
    AFrag a, b0, b1;
    const _Float16* src = (kb < NH) ? hs : hbk;
    int kk = (kb < NH) ? kb : (kb - NH);
    int kA = kk + (hi ? 8 : 0);
    a.q[0] = *(const u32x4*)(src + (size_t)arow * NH + kA);
    a.q[1] = *(const u32x4*)(src + (size_t)arow * NH + kA + 16);
    int kB = kb + (hi ? 8 : 0);
    const u32x4* bp0 = (const u32x4*)(wsh + (size_t)lnlo * 512 + kB);
    b0.q[0] = bp0[0]; b0.q[1] = bp0[2];
    const u32x4* bp1 = (const u32x4*)(wsh + (size_t)(16 + lnlo) * 512 + kB);
    b1.q[0] = bp1[0]; b1.q[1] = bp1[2];
    c0 = wmma16(a, b0, c0);
    c1 = wmma16(a, b1, c1);
  }
  for (int nt = 0; nt < 2; ++nt) {
    int n = nt * 16 + lnlo;
    if (n < NK) {
      float bn = bout[n];
      v8f c = nt ? c1 : c0;
      for (int r = 0; r < 8; ++r) {
        int m = mtile * 16 + r + hi * 8;
        int t = m / NB, b = m % NB;
        feats[((size_t)b * NT + t) * NK + n] = c[r] + bn;
      }
    }
  }
}

// ============ K5: Viterbi decode (thread per sequence) ============
__global__ void k_viterbi(const float* __restrict__ feats, const float* __restrict__ trans,
                          unsigned char* __restrict__ bp, float* __restrict__ out) {
  int b = threadIdx.x;
  if (b >= NB) return;
  float vit[NK];
  for (int k = 0; k < NK; ++k) vit[k] = NEGF;
  vit[START_TAG] = 0.0f;
  const float* fb = feats + (size_t)b * NT * NK;
  unsigned char* bpb = bp + (size_t)b * NT * NK;
  for (int t = 0; t < NT; ++t) {
    float nv[NK];
    for (int to = 0; to < NK; ++to) {
      float best = vit[0] + trans[to * NK + 0];
      int bi = 0;
      for (int fr = 1; fr < NK; ++fr) {
        float s = vit[fr] + trans[to * NK + fr];
        if (s > best) { best = s; bi = fr; }
      }
      bpb[t * NK + to] = (unsigned char)bi;
      nv[to] = best + fb[t * NK + to];
    }
    for (int k = 0; k < NK; ++k) vit[k] = nv[k];
  }
  float best = vit[0] + trans[STOP_TAG * NK + 0];
  int btag = 0;
  for (int k = 1; k < NK; ++k) {
    float s = vit[k] + trans[STOP_TAG * NK + k];
    if (s > best) { best = s; btag = k; }
  }
  out[b] = best;
  int tag = btag;
  for (int t = NT - 1; t >= 0; --t) {
    out[NB + (size_t)b * NT + t] = (float)tag;
    tag = bpb[t * NK + tag];
  }
}

// ================= host launcher =================
extern "C" void kernel_launch(void* const* d_in, const int* in_sizes, int n_in,
                              void* d_out, int out_size, void* d_ws, size_t ws_size,
                              hipStream_t stream) {
  const int*   sentences = (const int*)  d_in[0];
  const float* h0        = (const float*)d_in[1];
  const float* c0        = (const float*)d_in[2];
  const float* W_emb     = (const float*)d_in[3];
  const float* W_ih_f    = (const float*)d_in[4];
  const float* W_hh_f    = (const float*)d_in[5];
  const float* b_f       = (const float*)d_in[6];
  const float* W_ih_b    = (const float*)d_in[7];
  const float* W_hh_b    = (const float*)d_in[8];
  const float* b_b       = (const float*)d_in[9];
  const float* W_out     = (const float*)d_in[10];
  const float* b_out     = (const float*)d_in[11];
  const float* trans     = (const float*)d_in[12];
  float* out = (float*)d_out;

  // workspace layout (256B aligned regions)
  char* ws = (char*)d_ws;
  size_t off = 0;
  _Float16* x16    = (_Float16*)(ws + off); off += (size_t)TB * NEP * 2;          // 2.5 MB
  _Float16* wih16  = (_Float16*)(ws + off); off += (size_t)2 * G4 * NEP * 2;      // 1.25 MB
  _Float16* whh16  = (_Float16*)(ws + off); off += (size_t)2 * G4 * NH * 2;       // 1 MB
  _Float16* wout16 = (_Float16*)(ws + off); off += (size_t)32 * 512 * 2;          // 32 KB
  float*    pre2   = (float*)   (ws + off); off += (size_t)2 * NT * G4 * NB * 4;  // 32 MB
  _Float16* hs     = (_Float16*)(ws + off); off += (size_t)2 * NT * NB * NH * 2;  // 4 MB
  float*    feats  = (float*)   (ws + off); off += (size_t)NB * NT * NK * 4;      // 320 KB
  unsigned char* bp = (unsigned char*)(ws + off);                                  // 80 KB

  // K0: weight repack -> f16
  {
    int tot = 2 * G4 * NEP + 2 * G4 * NH + 32 * 512;
    k_convert<<<(tot + 255) / 256, 256, 0, stream>>>(W_ih_f, W_ih_b, W_hh_f, W_hh_b,
                                                     W_out, wih16, whh16, wout16);
  }
  // K1: embedding gather
  {
    int tot = TB * NEP;
    k_embed<<<(tot + 255) / 256, 256, 0, stream>>>(sentences, W_emb, x16);
  }
  // K2: input projection pre-gates (both directions), 16384 waves
  k_pregate<<<2048, 256, 0, stream>>>(x16, wih16, b_f, b_b, pre2);
  // K3: recurrence, one persistent block per direction
  k_lstm<<<2, 512, 0, stream>>>(pre2, whh16, h0, c0, hs);
  // K4: output projection
  k_feats<<<32, 256, 0, stream>>>(hs, wout16, b_out, feats);
  // K5: Viterbi decode + output write
  k_viterbi<<<1, 32, 0, stream>>>(feats, trans, bp, out);
}
